// Dhit2_23957327577884
// MI455X (gfx1250) — matrix operand order repaired and checked by reading, not compiled, not run on hardware
//
#include <hip/hip_runtime.h>
#include <hip/hip_fp16.h>
#include <math.h>

typedef _Float16 h16;
typedef _Float16 v16h __attribute__((ext_vector_type(16)));
typedef _Float16 h8v  __attribute__((ext_vector_type(8)));
typedef float    v8f  __attribute__((ext_vector_type(8)));

// ============================================================================
// Generic WMMA GEMM-NT:  C[M,N] = sum_w A_shift(w)[M,K] * B_w[N,K]^T  (+bias)
//   A : f16 row-major, lda;  rows shifted by (w - pad) per conv tap, OOB = 0
//   B : f16 row-major N x K (weights pre-transposed [w][cout][cin]), ldb
//   C : fp32, ldc.  gridDim.z batches with independent batch strides.
// Wave32: each wave owns a 16x32 C tile (two 16x16 accumulators sharing one A
// fragment), K consumed 32/instr via v_wmma_f32_16x16x32_f16.
// Hot loop: ping-pong double-buffered fragments (2x unrolled, no register
// rotation copies), unconditional b128 loads with partial s_wait_loadcnt,
// speculative global_prefetch_b8 on the K streams. Wave id scalarized via
// readfirstlane so path selection is a scalar branch. Edge tiles take a
// per-lane guarded path. A/B fragments use contiguous k = 16*(lane/16) + s order.
// ============================================================================
struct GemmP {
  const h16* A; long lda, aBatch;
  const h16* B; long ldb, bBatch, bSlice;
  const float* bias;
  float* C; long ldc, cBatch;
  int M, N, K, width, pad;
};

union Frag { v16h v; h8v h[2]; };

__device__ __forceinline__ v8f wmma16(h8v a0, h8v a1, h8v b0, h8v b1, v8f c) {
  Frag ua; ua.h[0] = a0; ua.h[1] = a1;
  Frag ub; ub.h[0] = b0; ub.h[1] = b1;
  return __builtin_amdgcn_wmma_f32_16x16x32_f16(
      false, ua.v, false, ub.v, (short)0, c, false, false);
}

struct Pipe { h8v a0, a1, b00, b01, b10, b11; };

__device__ __forceinline__ void loadPipe(Pipe& t, const h16* arp,
                                         const h16* brp0, const h16* brp1,
                                         int k0) {
  t.a0  = *(const h8v*)(arp + k0);
  t.a1  = *(const h8v*)(arp + k0 + 8);
  t.b00 = *(const h8v*)(brp0 + k0);
  t.b01 = *(const h8v*)(brp0 + k0 + 8);
  t.b10 = *(const h8v*)(brp1 + k0);
  t.b11 = *(const h8v*)(brp1 + k0 + 8);
}

__device__ __forceinline__ void accumPipe(const Pipe& t, v8f& acc0, v8f& acc1) {
  acc0 = wmma16(t.a0, t.a1, t.b00, t.b01, acc0);
  acc1 = wmma16(t.a0, t.a1, t.b10, t.b11, acc1);
}

__global__ __launch_bounds__(128) void k_gemm(GemmP p) {
  const int lane = threadIdx.x & 31;
  // scalarize the wave index -> mBase lives in SGPRs, full-tile tests become
  // scalar branches (no EXEC save/restore around the hot path)
  const int wv = __builtin_amdgcn_readfirstlane(threadIdx.x >> 5);
  const long mBase = (long)blockIdx.x * 64 + wv * 16;
  const int n0 = blockIdx.y * 32;

  const h16* A = p.A + (long)blockIdx.z * p.aBatch;
  const h16* B = p.B + (long)blockIdx.z * p.bBatch;
  float*     C = p.C + (long)blockIdx.z * p.cBatch;

  const int mrow  = lane & 15;                       // A: M index
  const int khalf = (lane >> 4) << 4;                // A: K sub-block 0/16
  const int koff  = (lane >> 4) << 4;                // B: K sub-block 0/16
  const int nr0   = n0 + (lane & 15);                // B: N index, sub-tile 0
  const int nr1   = nr0 + 16;                        // B: N index, sub-tile 1
  const bool nok0 = nr0 < p.N;
  const bool nok1 = nr1 < p.N;
  const bool nfull = (n0 + 32) <= p.N;               // block-uniform (scalar)

  v8f acc0 = {}, acc1 = {};

  for (int w = 0; w < p.width; ++w) {
    const long shift = (long)w - p.pad;
    const long ar = mBase + mrow + shift;            // conv row shift
    const bool aok = (ar >= 0) && (ar < (long)p.M);
    const bool afull = (mBase + shift >= 0) &&
                       (mBase + 15 + shift < (long)p.M);  // scalar (mBase SGPR)
    const h16* arp  = A + ar * p.lda + khalf;
    const h16* brp0 = B + (long)w * p.bSlice + (long)nr0 * p.ldb + koff;
    const h16* brp1 = B + (long)w * p.bSlice + (long)nr1 * p.ldb + koff;

    if (afull && nfull) {
      // ---- hot path: ping-pong double buffer, 2x unrolled ----
      Pipe E, O;
      loadPipe(E, arp, brp0, brp1, 0);
      const int nchunk = p.K >> 5;                   // K/32 (K % 32 == 0)
      int c = 1;
      for (; c + 1 < nchunk; c += 2) {
        const int k0 = c << 5;
        // speculative prefetch 512B ahead on each K stream
        // (global_prefetch_b8; invalid addresses silently dropped)
        __builtin_prefetch((const char*)(arp + k0) + 512, 0, 1);
        __builtin_prefetch((const char*)(brp0 + k0) + 512, 0, 1);
        __builtin_prefetch((const char*)(brp1 + k0) + 512, 0, 1);
        loadPipe(O, arp, brp0, brp1, k0);
        accumPipe(E, acc0, acc1);
        loadPipe(E, arp, brp0, brp1, k0 + 32);
        accumPipe(O, acc0, acc1);
      }
      if (c < nchunk) {                              // even chunk count tail
        loadPipe(O, arp, brp0, brp1, c << 5);
        accumPipe(E, acc0, acc1);
        accumPipe(O, acc0, acc1);
      } else {                                       // odd chunk count tail
        accumPipe(E, acc0, acc1);
      }
    } else {
      // ---- edge path: per-lane guarded, zero-filled fragments ----
      for (int k0 = 0; k0 < p.K; k0 += 32) {
        Pipe t;
        t.a0 = (h8v){}; t.a1 = (h8v){};
        t.b00 = (h8v){}; t.b01 = (h8v){};
        t.b10 = (h8v){}; t.b11 = (h8v){};
        if (aok) {
          t.a0 = *(const h8v*)(arp + k0);
          t.a1 = *(const h8v*)(arp + k0 + 8);
        }
        if (nok0) {
          t.b00 = *(const h8v*)(brp0 + k0);
          t.b01 = *(const h8v*)(brp0 + k0 + 8);
        }
        if (nok1) {
          t.b10 = *(const h8v*)(brp1 + k0);
          t.b11 = *(const h8v*)(brp1 + k0 + 8);
        }
        accumPipe(t, acc0, acc1);
      }
    }
  }

  // epilogue: C layout -> M = i + 8*(lane>=16), N = lane&15 (+16 for tile 1)
  const int mtop = (lane >> 4) << 3;
  if (nok0) {
    const float bv = p.bias ? p.bias[nr0] : 0.f;
#pragma unroll
    for (int i = 0; i < 8; ++i) {
      const long row = mBase + mtop + i;
      if (row < (long)p.M) C[row * p.ldc + nr0] = acc0[i] + bv;
    }
  }
  if (nok1) {
    const float bv = p.bias ? p.bias[nr1] : 0.f;
#pragma unroll
    for (int i = 0; i < 8; ++i) {
      const long row = mBase + mtop + i;
      if (row < (long)p.M) C[row * p.ldc + nr1] = acc1[i] + bv;
    }
  }
}

// ============================================================================
// Elementwise / small kernels (fp32 VALU path)
// ============================================================================
__device__ __forceinline__ float geluf(float v) {
  return v / (1.f + expf(-1.702f * v));
}

// stem conv: width 15, cin 4, direct (K=60 not a multiple of 32)
__global__ void k_stem(const float* __restrict__ x, const float* __restrict__ w,
                       const float* __restrict__ b, float* __restrict__ y,
                       int L, int cout) {
  long i = (long)blockIdx.x * blockDim.x + threadIdx.x;
  if (i >= (long)L * cout) return;
  int co = (int)(i % cout); long l = i / cout;
  float s = b[co];
  for (int kw = 0; kw < 15; ++kw) {
    long r = l + kw - 7;
    if (r < 0 || r >= L) continue;
#pragma unroll
    for (int ci = 0; ci < 4; ++ci)
      s += x[r * 4 + ci] * w[(kw * 4 + ci) * (long)cout + co];
  }
  y[i] = s;
}

__global__ void k_bn_gelu(const float* __restrict__ x, const float* g,
                          const float* bt, const float* mn, const float* vr,
                          h16* __restrict__ y, long n, int c) {
  long i = (long)blockIdx.x * blockDim.x + threadIdx.x;
  if (i >= n) return;
  int ch = (int)(i % c);
  float v = (x[i] - mn[ch]) * rsqrtf(vr[ch] + 1e-3f) * g[ch] + bt[ch];
  y[i] = (h16)geluf(v);
}

__global__ void k_gelu(float* x, long n) {
  long i = (long)blockIdx.x * blockDim.x + threadIdx.x;
  if (i < n) x[i] = geluf(x[i]);
}

__global__ void k_relu_h(const float* x, h16* y, long n) {
  long i = (long)blockIdx.x * blockDim.x + threadIdx.x;
  if (i < n) y[i] = (h16)fmaxf(x[i], 0.f);
}

__global__ void k_add(float* y, const float* x, long n) {
  long i = (long)blockIdx.x * blockDim.x + threadIdx.x;
  if (i < n) y[i] += x[i];
}

__global__ void k_pool(const float* x, float* y, long halfL, int c) {
  long i = (long)blockIdx.x * blockDim.x + threadIdx.x;
  if (i >= halfL * c) return;
  int ch = (int)(i % c); long l = i / c;
  y[i] = fmaxf(x[(2 * l) * c + ch], x[(2 * l + 1) * c + ch]);
}

// weight transpose+convert: w[width][cin][cout] f32 -> wt[width][cout][cin] f16
__global__ void k_wt(const float* __restrict__ w, h16* __restrict__ wt,
                     int width, int cin, int cout) {
  long i = (long)blockIdx.x * blockDim.x + threadIdx.x;
  if (i >= (long)width * cin * cout) return;
  int ci = (int)(i % cin); long t = i / cin;
  int co = (int)(t % cout); int wd = (int)(t / cout);
  wt[i] = (h16)w[((long)wd * cin + ci) * cout + co];
}

// 2D convert f32 -> f16 (dst may be col-offset for concat)
__global__ void k_cvt2d(const float* __restrict__ s, long sld,
                        h16* __restrict__ d, long dld, long rows, long cols) {
  long i = (long)blockIdx.x * blockDim.x + threadIdx.x;
  if (i >= rows * cols) return;
  long r = i / cols, c = i % cols;
  d[r * dld + c] = (h16)s[r * sld + c];
}

// v (T x 768) -> vt f16 [H][VS][T]
__global__ void k_vt(const float* __restrict__ v, h16* __restrict__ vt, int T) {
  long i = (long)blockIdx.x * blockDim.x + threadIdx.x;
  if (i >= (long)768 * T) return;
  int t = (int)(i % T); long c = i / T;  // c = h*96+vd
  vt[i] = (h16)v[(long)t * 768 + c];
}

// q scaled + r_w / r_r biases -> f16 A operands
__global__ void k_qbias(const float* __restrict__ q, const float* rw,
                        const float* rr, h16* qw, h16* qr, long n) {
  long i = (long)blockIdx.x * blockDim.x + threadIdx.x;
  if (i >= n) return;
  int col = (int)(i % 512);
  float v = q[i] * 0.125f;  // KS^-0.5
  qw[i] = (h16)(v + rw[col]);
  qr[i] = (h16)(v + rr[col]);
}

// layernorm (eps 1e-3), row = 768, out f16
__global__ void k_ln(const float* __restrict__ x, const float* g,
                     const float* b, h16* __restrict__ y) {
  const int C = 768, tid = threadIdx.x;
  const float* xr = x + (long)blockIdx.x * C;
  h16* yr = y + (long)blockIdx.x * C;
  __shared__ float red[256];
  float s = 0.f;
  for (int i = tid; i < C; i += 256) s += xr[i];
  red[tid] = s; __syncthreads();
  for (int o = 128; o; o >>= 1) { if (tid < o) red[tid] += red[tid + o]; __syncthreads(); }
  float mu = red[0] / C; __syncthreads();
  float vs = 0.f;
  for (int i = tid; i < C; i += 256) { float d = xr[i] - mu; vs += d * d; }
  red[tid] = vs; __syncthreads();
  for (int o = 128; o; o >>= 1) { if (tid < o) red[tid] += red[tid + o]; __syncthreads(); }
  float inv = rsqrtf(red[0] / C + 1e-3f);
  for (int i = tid; i < C; i += 256) yr[i] = (h16)((xr[i] - mu) * inv * g[i] + b[i]);
}

// softmax over s with analytic relative_shift: rel row = T-1+s-t
__global__ void k_softmax(const float* __restrict__ sc, const float* __restrict__ sr,
                          h16* __restrict__ attn) {
  const int T = 256;
  int t = blockIdx.x, h = blockIdx.y, s = threadIdx.x;
  long base = ((long)h * T + t);
  float v = sc[base * T + s] + sr[base * (2 * T - 1) + (T - 1 + s - t)];
  __shared__ float red[256];
  red[s] = v; __syncthreads();
  for (int o = 128; o; o >>= 1) { if (s < o) red[s] = fmaxf(red[s], red[s + o]); __syncthreads(); }
  float mx = red[0]; __syncthreads();
  float e = expf(v - mx);
  red[s] = e; __syncthreads();
  for (int o = 128; o; o >>= 1) { if (s < o) red[s] += red[s + o]; __syncthreads(); }
  attn[base * T + s] = (h16)(e / red[0]);
}

// positional features: one block per basis column b in [0,16)
__global__ void k_pos(float* __restrict__ pe) {
  const int bcol = blockIdx.x, tid = threadIdx.x;
  const float hl   = exp2f(3.f + 5.f * (float)bcol / 15.f);       // 2^linspace(3,8,16)
  const float cw   = exp2f((float)(bcol + 1)) - 1.f;
  const float sd   = 8.f;                                         // 256/(2*16)
  const float mean = 16.f + 240.f * (float)bcol / 15.f;           // linspace(16,256,16)
  const float conc = (mean / sd) * (mean / sd);
  const float rate = mean / (sd * sd);
  const float nrm  = lgammaf(conc) - conc * logf(rate);
  __shared__ float red[256];
  float mx = 0.f;
  for (int r = tid; r < 511; r += 256) {
    float ap = fabsf((float)(r - 255));
    float pdf = expf((conc - 1.f) * logf(ap) - rate * ap - nrm) + 1e-8f;
    mx = fmaxf(mx, pdf);
  }
  red[tid] = mx; __syncthreads();
  for (int o = 128; o; o >>= 1) { if (tid < o) red[tid] = fmaxf(red[tid], red[tid + o]); __syncthreads(); }
  const float pmax = red[0]; __syncthreads();
  const float ln2 = 0.6931471805599453f;
  for (int r = tid; r < 511; r += 256) {
    float pos = (float)(r - 255), ap = fabsf(pos);
    float fe = expf(-ln2 / hl * ap);
    float fc = (cw > ap) ? 1.f : 0.f;
    float fg = (expf((conc - 1.f) * logf(ap) - rate * ap - nrm) + 1e-8f) / pmax;
    float sg = (pos > 0.f) ? 1.f : ((pos < 0.f) ? -1.f : 0.f);
    float* row = pe + (long)r * 96;
    row[bcol] = fe; row[16 + bcol] = fc; row[32 + bcol] = fg;
    row[48 + bcol] = sg * fe; row[64 + bcol] = sg * fc; row[80 + bcol] = sg * fg;
  }
}

// ============================================================================
// Host orchestration
// ============================================================================
static inline dim3 g1(long n) { return dim3((unsigned)((n + 255) / 256)); }

static void gemm(hipStream_t st, const h16* A, long lda, long aB,
                 const h16* B, long ldb, long bB, long bSlice,
                 const float* bias, float* C, long ldc, long cB,
                 int M, int N, int K, int width, int pad, int batch) {
  GemmP p{A, lda, aB, B, ldb, bB, bSlice, bias, C, ldc, cB, M, N, K, width, pad};
  dim3 g((unsigned)((M + 63) / 64), (unsigned)((N + 31) / 32), (unsigned)batch);
  k_gemm<<<g, 128, 0, st>>>(p);
}

// cb_p leaves (insertion order): gamma,beta,mean,var,w,b
struct CB { const float *g, *bt, *mn, *vr, *w, *b; };
static CB cb(void* const* d, int i) {
  return CB{(const float*)d[i], (const float*)d[i+1], (const float*)d[i+2],
            (const float*)d[i+3], (const float*)d[i+4], (const float*)d[i+5]};
}

// stem_tower_p leaves: stem_w, stem_b, stem_res(cb:6), tower 6x{cb:6, res:6} = 80
static void run_tower(hipStream_t st, const float* x, void* const* d, int base,
                      float* F0, float* F1, h16* H16, h16* WT) {
  const int FIL[6] = {384, 448, 512, 576, 640, 768};
  // stem conv (32768 x 4 -> 32768 x 384)
  k_stem<<<g1(32768L * 384), 256, 0, st>>>(x, (const float*)d[base],
                                           (const float*)d[base + 1], F0, 32768, 384);
  // stem residual conv_block (width 1)
  CB sr = cb(d, base + 2);
  k_bn_gelu<<<g1(32768L * 384), 256, 0, st>>>(F0, sr.g, sr.bt, sr.mn, sr.vr, H16, 32768L * 384, 384);
  k_wt<<<g1(384L * 384), 256, 0, st>>>(sr.w, WT, 1, 384, 384);
  gemm(st, H16, 384, 0, WT, 384, 0, 0, sr.b, F1, 384, 0, 32768, 384, 384, 1, 0, 1);
  k_add<<<g1(32768L * 384), 256, 0, st>>>(F0, F1, 32768L * 384);
  k_pool<<<g1(16384L * 384), 256, 0, st>>>(F0, F1, 16384, 384);

  long L = 16384; int cin = 384;
  float* cur = F1; float* oth = F0;
  for (int j = 0; j < 6; ++j) {
    int nf = FIL[j];
    CB cp = cb(d, base + 8 + 12 * j);
    CB rp = cb(d, base + 8 + 12 * j + 6);
    k_bn_gelu<<<g1(L * cin), 256, 0, st>>>(cur, cp.g, cp.bt, cp.mn, cp.vr, H16, L * cin, cin);
    k_wt<<<g1(5L * cin * nf), 256, 0, st>>>(cp.w, WT, 5, cin, nf);
    gemm(st, H16, cin, 0, WT, cin, 0, (long)nf * cin, cp.b, oth, nf, 0,
         (int)L, nf, cin, 5, 2, 1);
    k_bn_gelu<<<g1(L * nf), 256, 0, st>>>(oth, rp.g, rp.bt, rp.mn, rp.vr, H16, L * nf, nf);
    k_wt<<<g1((long)nf * nf), 256, 0, st>>>(rp.w, WT, 1, nf, nf);
    gemm(st, H16, nf, 0, WT, nf, 0, 0, rp.b, cur, nf, 0, (int)L, nf, nf, 1, 0, 1);
    k_add<<<g1(L * nf), 256, 0, st>>>(oth, cur, L * nf);
    k_pool<<<g1((L / 2) * nf), 256, 0, st>>>(oth, cur, L / 2, nf);
    L /= 2; cin = nf;
  }
  // result: cur == F1, 256 x 768
}

extern "C" void kernel_launch(void* const* d_in, const int* in_sizes, int n_in,
                              void* d_out, int out_size, void* d_ws, size_t ws_size,
                              hipStream_t stream) {
  (void)in_sizes; (void)n_in; (void)out_size; (void)ws_size;
  // ---- leaf index map (insertion-order flatten of setup_inputs dict) ----
  // 0:x 1:x_seq | conv:2..81 conv_seq:82..161 | 162:d_in_w 163:d_in_b
  // blocks: 164+16*i  {ln1_g,ln1_b,Wq,Wk,Wv,Wr,r_w,r_r,Wo,bo,ln2_g,ln2_b,W1,b1,W2,b2}
  // 308:d_mid_w 309:d_mid_b | fp1:310 fp2:316 fp3:322 | 328:d_out_w 329:d_out_b
  auto P = [&](int i) { return (const float*)d_in[i]; };

  size_t off = 0;
  auto carve = [&](size_t bytes) {
    char* p = (char*)d_ws + off;
    off = (off + bytes + 255) & ~(size_t)255;
    return (void*)p;
  };
  h16*   WT    = (h16*)  carve(5242880ull * 2);   // f16 weight staging (max 1536*3072)
  float* F0    = (float*)carve(12582912ull * 4);  // act ping
  float* F1    = (float*)carve(12582912ull * 4);  // act pong
  h16*   H16   = (h16*)  carve(12582912ull * 2);  // f16 act staging
  h16*   hcat  = (h16*)  carve(256ull * 1536 * 2);
  float* hF    = (float*)carve(256ull * 768 * 4);
  h16*   xnH   = (h16*)  carve(256ull * 768 * 2);
  float* qF    = (float*)carve(256ull * 512 * 4);
  float* kF    = (float*)carve(256ull * 512 * 4);
  h16*   kH    = (h16*)  carve(256ull * 512 * 2);
  float* vF    = (float*)carve(256ull * 768 * 4);
  h16*   vtH   = (h16*)  carve(768ull * 256 * 2);
  h16*   qwH   = (h16*)  carve(256ull * 512 * 2);
  h16*   qrH   = (h16*)  carve(256ull * 512 * 2);
  float* posF  = (float*)carve(511ull * 96 * 4);
  h16*   posH  = (h16*)  carve(511ull * 96 * 2);
  float* relkF = (float*)carve(511ull * 512 * 4);
  h16*   relkH = (h16*)  carve(511ull * 512 * 2);
  float* scF   = (float*)carve(8ull * 256 * 256 * 4);
  float* srF   = (float*)carve(8ull * 256 * 511 * 4);
  h16*   attnH = (h16*)  carve(8ull * 256 * 256 * 2);
  float* oF    = (float*)carve(256ull * 768 * 4);
  h16*   oH    = (h16*)  carve(256ull * 768 * 2);
  float* ffF   = (float*)carve(256ull * 1536 * 4);
  h16*   ffH   = (h16*)  carve(256ull * 1536 * 2);
  float* aoF   = (float*)carve(256ull * 768 * 4);
  float* hmF   = (float*)carve(192ull * 768 * 4);
  h16*   gH    = (h16*)  carve(192ull * 3072 * 2);
  float* s1F   = (float*)carve(192ull * 3072 * 4);
  float* s2F   = (float*)carve(192ull * 3072 * 4);
  h16*   hhH   = (h16*)  carve(256ull * 768 * 2);

  // ---- conv towers ----
  run_tower(stream, P(0), d_in, 2,  F0, F1, H16, WT);
  k_cvt2d<<<g1(256L * 768), 256, 0, stream>>>(F1, 768, hcat, 1536, 256, 768);
  run_tower(stream, P(1), d_in, 82, F0, F1, H16, WT);
  k_cvt2d<<<g1(256L * 768), 256, 0, stream>>>(F1, 768, hcat + 768, 1536, 256, 768);

  // ---- d_in projection ----
  k_wt<<<g1(1536L * 768), 256, 0, stream>>>(P(162), WT, 1, 1536, 768);
  gemm(stream, hcat, 1536, 0, WT, 1536, 0, 0, P(163), hF, 768, 0, 256, 768, 1536, 1, 0, 1);

  // ---- positional features ----
  k_pos<<<16, 256, 0, stream>>>(posF);
  k_cvt2d<<<g1(511L * 96), 256, 0, stream>>>(posF, 96, posH, 96, 511, 96);

  // ---- transformer blocks ----
  for (int blk = 0; blk < 9; ++blk) {
    int pb = 164 + 16 * blk;
    k_ln<<<256, 256, 0, stream>>>(hF, P(pb + 0), P(pb + 1), xnH);
    // Q, K, V projections
    k_wt<<<g1(768L * 512), 256, 0, stream>>>(P(pb + 2), WT, 1, 768, 512);
    gemm(stream, xnH, 768, 0, WT, 768, 0, 0, nullptr, qF, 512, 0, 256, 512, 768, 1, 0, 1);
    k_wt<<<g1(768L * 512), 256, 0, stream>>>(P(pb + 3), WT, 1, 768, 512);
    gemm(stream, xnH, 768, 0, WT, 768, 0, 0, nullptr, kF, 512, 0, 256, 512, 768, 1, 0, 1);
    k_cvt2d<<<g1(256L * 512), 256, 0, stream>>>(kF, 512, kH, 512, 256, 512);
    k_wt<<<g1(768L * 768), 256, 0, stream>>>(P(pb + 4), WT, 1, 768, 768);
    gemm(stream, xnH, 768, 0, WT, 768, 0, 0, nullptr, vF, 768, 0, 256, 768, 768, 1, 0, 1);
    k_vt<<<g1(768L * 256), 256, 0, stream>>>(vF, vtH, 256);
    // rel_k = pos_emb @ Wr
    k_wt<<<g1(96L * 512), 256, 0, stream>>>(P(pb + 5), WT, 1, 96, 512);
    gemm(stream, posH, 96, 0, WT, 96, 0, 0, nullptr, relkF, 512, 0, 511, 512, 96, 1, 0, 1);
    k_cvt2d<<<g1(511L * 512), 256, 0, stream>>>(relkF, 512, relkH, 512, 511, 512);
    // q*scale + r_w / r_r
    k_qbias<<<g1(256L * 512), 256, 0, stream>>>(qF, P(pb + 6), P(pb + 7), qwH, qrH, 256L * 512);
    // scores: content (8 heads batched), relative
    gemm(stream, qwH, 512, 64, kH, 512, 64, 0, nullptr, scF, 256, 256L * 256,
         256, 256, 64, 1, 0, 8);
    gemm(stream, qrH, 512, 64, relkH, 512, 64, 0, nullptr, srF, 511, 256L * 511,
         256, 511, 64, 1, 0, 8);
    k_softmax<<<dim3(256, 8), 256, 0, stream>>>(scF, srF, attnH);
    // out = attn @ v  (NT with vt[H][VS][T])
    gemm(stream, attnH, 256, 256L * 256, vtH, 256, 96L * 256, 0, nullptr,
         oF, 768, 96, 256, 96, 256, 1, 0, 8);
    k_cvt2d<<<g1(256L * 768), 256, 0, stream>>>(oF, 768, oH, 768, 256, 768);
    k_wt<<<g1(768L * 768), 256, 0, stream>>>(P(pb + 8), WT, 1, 768, 768);
    gemm(stream, oH, 768, 0, WT, 768, 0, 0, P(pb + 9), aoF, 768, 0, 256, 768, 768, 1, 0, 1);
    k_add<<<g1(256L * 768), 256, 0, stream>>>(hF, aoF, 256L * 768);
    // FFN
    k_ln<<<256, 256, 0, stream>>>(hF, P(pb + 10), P(pb + 11), xnH);
    k_wt<<<g1(768L * 1536), 256, 0, stream>>>(P(pb + 12), WT, 1, 768, 1536);
    gemm(stream, xnH, 768, 0, WT, 768, 0, 0, P(pb + 13), ffF, 1536, 0, 256, 1536, 768, 1, 0, 1);
    k_relu_h<<<g1(256L * 1536), 256, 0, stream>>>(ffF, ffH, 256L * 1536);
    k_wt<<<g1(1536L * 768), 256, 0, stream>>>(P(pb + 14), WT, 1, 1536, 768);
    gemm(stream, ffH, 1536, 0, WT, 1536, 0, 0, P(pb + 15), aoF, 768, 0, 256, 768, 1536, 1, 0, 1);
    k_add<<<g1(256L * 768), 256, 0, stream>>>(hF, aoF, 256L * 768);
  }

  // ---- head: trim 32..223, d_mid, fp1..fp3, d_out ----
  k_cvt2d<<<g1(256L * 768), 256, 0, stream>>>(hF, 768, hhH, 768, 256, 768);
  k_wt<<<g1(768L * 768), 256, 0, stream>>>(P(308), WT, 1, 768, 768);
  gemm(stream, hhH + 32L * 768, 768, 0, WT, 768, 0, 0, P(309), hmF, 768, 0,
       192, 768, 768, 1, 0, 1);
  CB f1 = cb(d_in, 310), f2 = cb(d_in, 316), f3 = cb(d_in, 322);
  // fp1
  k_bn_gelu<<<g1(192L * 768), 256, 0, stream>>>(hmF, f1.g, f1.bt, f1.mn, f1.vr, gH, 192L * 768, 768);
  k_wt<<<g1(768L * 1536), 256, 0, stream>>>(f1.w, WT, 1, 768, 1536);
  gemm(stream, gH, 768, 0, WT, 768, 0, 0, f1.b, s1F, 1536, 0, 192, 1536, 768, 1, 0, 1);
  k_gelu<<<g1(192L * 1536), 256, 0, stream>>>(s1F, 192L * 1536);
  // fp2
  k_bn_gelu<<<g1(192L * 1536), 256, 0, stream>>>(s1F, f2.g, f2.bt, f2.mn, f2.vr, gH, 192L * 1536, 1536);
  k_wt<<<g1(1536L * 3072), 256, 0, stream>>>(f2.w, WT, 1, 1536, 3072);
  gemm(stream, gH, 1536, 0, WT, 1536, 0, 0, f2.b, s2F, 3072, 0, 192, 3072, 1536, 1, 0, 1);
  k_gelu<<<g1(192L * 3072), 256, 0, stream>>>(s2F, 192L * 3072);
  // fp3
  k_bn_gelu<<<g1(192L * 3072), 256, 0, stream>>>(s2F, f3.g, f3.bt, f3.mn, f3.vr, gH, 192L * 3072, 3072);
  k_wt<<<g1(3072L * 1536), 256, 0, stream>>>(f3.w, WT, 1, 3072, 1536);
  gemm(stream, gH, 3072, 0, WT, 3072, 0, 0, f3.b, s1F, 1536, 0, 192, 1536, 3072, 1, 0, 1);
  k_gelu<<<g1(192L * 1536), 256, 0, stream>>>(s1F, 192L * 1536);
  // d_out -> (192, 73) fp32 directly into d_out
  k_cvt2d<<<g1(192L * 1536), 256, 0, stream>>>(s1F, 1536, gH, 1536, 192, 1536);
  k_wt<<<g1(1536L * 73), 256, 0, stream>>>(P(328), WT, 1, 1536, 73);
  gemm(stream, gH, 1536, 0, WT, 1536, 0, 0, P(329), (float*)d_out, 73, 0,
       192, 73, 1536, 1, 0, 1);
}
